// BahdanauAttention_704374636966
// MI455X (gfx1250) — matrix operand order repaired and checked by reading, not compiled, not run on hardware
//
#include <hip/hip_runtime.h>
#include <hip/hip_bf16.h>
#include <math.h>

// Problem dims (fixed by the reference)
#define B_  32
#define S_  2048
#define H_  1024
#define MT  64          // keys rows per workgroup in the WMMA scores kernel
#define KS  1032        // padded LDS row stride in bf16 elems (2064B = 516 dwords, 516%64=4 -> no bank conflicts)

typedef __attribute__((ext_vector_type(16))) __bf16        v16bf;
typedef __attribute__((ext_vector_type(8)))  float         v8f;
typedef __attribute__((ext_vector_type(4)))  unsigned int  u32x4;
typedef __attribute__((ext_vector_type(4)))  float         f32x4;

union Frag16 { v16bf v; u32x4 q[2]; };   // 16 bf16 per lane = 8 VGPRs

__device__ __forceinline__ float fast_tanh(float x) {
    x = fminf(fmaxf(x, -15.f), 15.f);       // avoid inf/inf
    float e = __expf(2.f * x);              // v_exp_f32 (TRANS pipe)
    return (e - 1.f) / (e + 1.f);
}

// ---------------------------------------------------------------------------
// Kernel 1: Ua_w fp32 -> bf16 (2 MB, stays resident in 192 MB L2)
// ---------------------------------------------------------------------------
__global__ __launch_bounds__(256) void cvt_ua_bf16(const float* __restrict__ Ua,
                                                   __bf16* __restrict__ UaB) {
    int i = (blockIdx.x * 256 + threadIdx.x) * 4;
    f32x4 f = *(const f32x4*)(Ua + i);
    __bf16* d = UaB + i;
    d[0] = (__bf16)f.x; d[1] = (__bf16)f.y; d[2] = (__bf16)f.z; d[3] = (__bf16)f.w;
}

// ---------------------------------------------------------------------------
// Kernel 2: t[b,g] = sum_h query[b,h]*Wa_w[g,h] + Wa_b[g] + Ua_b[g]   (tiny)
// ---------------------------------------------------------------------------
__global__ __launch_bounds__(256) void qproj_kernel(const float* __restrict__ query,
                                                    const float* __restrict__ Wa_w,
                                                    const float* __restrict__ Wa_b,
                                                    const float* __restrict__ Ua_b,
                                                    float* __restrict__ tQ) {
    __shared__ float q[H_];
    const int b = blockIdx.y;
    const int g = blockIdx.x * 256 + threadIdx.x;
    for (int h = threadIdx.x; h < H_; h += 256) q[h] = query[b * H_ + h];
    __syncthreads();
    float acc = Wa_b[g] + Ua_b[g];
    const float* wrow = Wa_w + (size_t)g * H_;
    for (int h = 0; h < H_; h += 4) {
        f32x4 w = *(const f32x4*)(wrow + h);
        acc = fmaf(q[h+0], w.x, acc);
        acc = fmaf(q[h+1], w.y, acc);
        acc = fmaf(q[h+2], w.z, acc);
        acc = fmaf(q[h+3], w.w, acc);
    }
    tQ[b * H_ + g] = acc;
}

// ---------------------------------------------------------------------------
// Kernel 3 (the hot one): fused  scores[b,s] = Va . tanh(t[b] + Ua_w @ keys[b,s]) + Va_b
// v_wmma_f32_16x16x32_bf16, 2 N-chunk register blocking, two-stage parity
// double-buffered K loop (no register rotation -> no v_mov / hazard NOPs).
// Grid: (S_/MT, B_), 128 threads (4 waves, one 16-row M-subtile each).
// Dynamic LDS: MT x KS bf16 keys tile (~129 KB of the WGP's 320 KB).
// ---------------------------------------------------------------------------
__global__ __launch_bounds__(128) void scores_kernel(const float*  __restrict__ keys,
                                                     const __bf16* __restrict__ UaB,
                                                     const float*  __restrict__ tQ,
                                                     const float*  __restrict__ Va,
                                                     const float*  __restrict__ Vab,
                                                     float* __restrict__ scores) {
    extern __shared__ char smem_raw[];
    __bf16* ktile = (__bf16*)smem_raw;

    const int b   = blockIdx.y;
    const int s0  = blockIdx.x * MT;
    const int tid = threadIdx.x;

    // Stage keys[b, s0:s0+MT, :] -> LDS, converting fp32 -> bf16 (coalesced b128 loads)
    const float* kb = keys + ((size_t)b * S_ + s0) * H_;
    for (int idx = tid; idx < MT * (H_ / 4); idx += 128) {
        const int row = idx >> 8;             // H_/4 == 256 float4 per row
        const int col = (idx & 255) * 4;
        f32x4 f = *(const f32x4*)(kb + (size_t)row * H_ + col);
        __bf16* d = ktile + row * KS + col;
        d[0] = (__bf16)f.x; d[1] = (__bf16)f.y; d[2] = (__bf16)f.z; d[3] = (__bf16)f.w;
    }
    __syncthreads();

    const int wv    = tid >> 5;       // wave id: owns M-subtile rows [wv*16, wv*16+16)
    const int lane  = tid & 31;
    const int l15   = lane & 15;
    const int lhalf = lane >> 4;

    // A-fragment base (lane = row, halves hold K[0..15] / K[16..31], matching B)
    const __bf16* ap0 = ktile + (wv * 16 + l15) * KS + lhalf * 16;

    float rowacc[8];
    #pragma unroll
    for (int r = 0; r < 8; ++r) rowacc[r] = 0.f;

    // N-loop: two 16-column chunks per iteration (register blocking).
    for (int n0 = 0; n0 < H_; n0 += 32) {
        v8f c0 = {};
        v8f c1 = {};
        // B element (k,n) = Ua_w[n0+n, k]; per lane: 16 contiguous K of one Ua_w row
        const __bf16* bp0 = UaB + (size_t)(n0 + l15) * H_ + lhalf * 16;
        const __bf16* bp1 = bp0 + (size_t)16 * H_;

        // two fragment stages, loaded and consumed in place (no rotation copies)
        Frag16 aA, b0A, b1A, aB, b0B, b1B;

        auto loadStage = [&](Frag16& a, Frag16& f0, Frag16& f1, int k) {
            a.q[0]  = *(const u32x4*)(ap0 + k);
            a.q[1]  = *(const u32x4*)(ap0 + k + 8);
            f0.q[0] = *(const u32x4*)(bp0 + k);
            f0.q[1] = *(const u32x4*)(bp0 + k + 8);
            f1.q[0] = *(const u32x4*)(bp1 + k);
            f1.q[1] = *(const u32x4*)(bp1 + k + 8);
        };

        loadStage(aA, b0A, b1A, 0);                    // prologue

        #pragma unroll 4
        for (int k0 = 0; k0 < H_; k0 += 64) {          // two 32-K steps per iteration
            loadStage(aB, b0B, b1B, k0 + 32);          // stage B loads fly over stage A WMMAs
            c0 = __builtin_amdgcn_wmma_f32_16x16x32_bf16(false, aA.v, false, b0A.v,
                                                         (short)0, c0, false, false);
            c1 = __builtin_amdgcn_wmma_f32_16x16x32_bf16(false, aA.v, false, b1A.v,
                                                         (short)0, c1, false, false);
            if (k0 + 64 < H_)                          // stage A loads fly over stage B WMMAs
                loadStage(aA, b0A, b1A, k0 + 64);
            c0 = __builtin_amdgcn_wmma_f32_16x16x32_bf16(false, aB.v, false, b0B.v,
                                                         (short)0, c0, false, false);
            c1 = __builtin_amdgcn_wmma_f32_16x16x32_bf16(false, aB.v, false, b1B.v,
                                                         (short)0, c1, false, false);
        }

        // C layout: VGPR r -> M = r + 8*lhalf, N = l15.  Fuse +t, tanh, *Va, row-accumulate.
        const float tv0 = tQ[b * H_ + n0 + l15];
        const float vv0 = Va[n0 + l15];
        const float tv1 = tQ[b * H_ + n0 + 16 + l15];
        const float vv1 = Va[n0 + 16 + l15];
        #pragma unroll
        for (int r = 0; r < 8; ++r) {
            rowacc[r] += fast_tanh(c0[r] + tv0) * vv0;
            rowacc[r] += fast_tanh(c1[r] + tv1) * vv1;
        }
    }

    // Reduce the 16 lanes (different N) that share each row; halves hold distinct rows.
    #pragma unroll
    for (int r = 0; r < 8; ++r) {
        float v = rowacc[r];
        v += __shfl_xor(v, 1, 32);
        v += __shfl_xor(v, 2, 32);
        v += __shfl_xor(v, 4, 32);
        v += __shfl_xor(v, 8, 32);
        rowacc[r] = v;
    }
    if (l15 == 0) {
        const float vb = Vab[0];
        #pragma unroll
        for (int r = 0; r < 8; ++r) {
            const int m = wv * 16 + r + lhalf * 8;
            scores[(size_t)b * S_ + s0 + m] = rowacc[r] + vb;
        }
    }
}

// ---------------------------------------------------------------------------
// Kernel 4: per-batch softmax over S + context = sum_s w[s]*keys[b,s,:]
// Grid: B_, 256 threads. Bandwidth bound (re-reads keys once, 256 MB total).
// ---------------------------------------------------------------------------
__global__ __launch_bounds__(256) void softmax_ctx_kernel(const float* __restrict__ keys,
                                                          const float* __restrict__ scores,
                                                          float* __restrict__ outC,
                                                          float* __restrict__ outW) {
    __shared__ float wbuf[S_];
    __shared__ float red[256];
    const int b   = blockIdx.x;
    const int tid = threadIdx.x;

    float lmax = -INFINITY;
    for (int s = tid; s < S_; s += 256) {
        float v = scores[(size_t)b * S_ + s];
        wbuf[s] = v;
        lmax = fmaxf(lmax, v);
    }
    red[tid] = lmax; __syncthreads();
    for (int off = 128; off > 0; off >>= 1) {
        if (tid < off) red[tid] = fmaxf(red[tid], red[tid + off]);
        __syncthreads();
    }
    const float gmax = red[0]; __syncthreads();

    float lsum = 0.f;
    for (int s = tid; s < S_; s += 256) {
        float e = __expf(wbuf[s] - gmax);
        wbuf[s] = e;
        lsum += e;
    }
    red[tid] = lsum; __syncthreads();
    for (int off = 128; off > 0; off >>= 1) {
        if (tid < off) red[tid] += red[tid + off];
        __syncthreads();
    }
    const float inv = 1.f / red[0]; __syncthreads();

    for (int s = tid; s < S_; s += 256) {
        wbuf[s] *= inv;
        outW[(size_t)b * S_ + s] = wbuf[s];
    }
    __syncthreads();

    // context[b, h4..h4+3]; coalesced float4 across the block (4 KB per s-row)
    const int h = tid * 4;
    f32x4 acc = {0.f, 0.f, 0.f, 0.f};
    const float* kb = keys + (size_t)b * S_ * H_;
    for (int s = 0; s < S_; ++s) {
        const float w = wbuf[s];
        f32x4 k = *(const f32x4*)(kb + (size_t)s * H_ + h);
        acc.x = fmaf(w, k.x, acc.x);
        acc.y = fmaf(w, k.y, acc.y);
        acc.z = fmaf(w, k.z, acc.z);
        acc.w = fmaf(w, k.w, acc.w);
    }
    *(f32x4*)(outC + (size_t)b * H_ + h) = acc;
}

// ---------------------------------------------------------------------------
extern "C" void kernel_launch(void* const* d_in, const int* in_sizes, int n_in,
                              void* d_out, int out_size, void* d_ws, size_t ws_size,
                              hipStream_t stream) {
    const float* query = (const float*)d_in[0];   // [B,H]
    const float* keys  = (const float*)d_in[1];   // [B,S,H]
    const float* Wa_w  = (const float*)d_in[2];   // [H,H]
    const float* Wa_b  = (const float*)d_in[3];   // [H]
    const float* Ua_w  = (const float*)d_in[4];   // [H,H]
    const float* Ua_b  = (const float*)d_in[5];   // [H]
    const float* Va_w  = (const float*)d_in[6];   // [1,H]
    const float* Va_b  = (const float*)d_in[7];   // [1]

    float* outC = (float*)d_out;                  // context [B,1,H] flat
    float* outW = (float*)d_out + (size_t)B_ * H_;// weights [B,S]   flat

    // workspace layout: UaB (2 MB) | tQ (128 KB) | scores (256 KB)
    char*   ws     = (char*)d_ws;
    __bf16* UaB    = (__bf16*)ws;
    float*  tQ     = (float*)(ws + (size_t)H_ * H_ * 2);
    float*  scores = (float*)(ws + (size_t)H_ * H_ * 2 + (size_t)B_ * H_ * 4);

    cvt_ua_bf16<<<dim3(H_ * H_ / (256 * 4)), dim3(256), 0, stream>>>(Ua_w, UaB);
    qproj_kernel<<<dim3(H_ / 256, B_), dim3(256), 0, stream>>>(query, Wa_w, Wa_b, Ua_b, tQ);

    const size_t lds_bytes = (size_t)MT * KS * sizeof(__bf16);  // ~129 KB dynamic LDS
    scores_kernel<<<dim3(S_ / MT, B_), dim3(128), lds_bytes, stream>>>(
        keys, UaB, tQ, Va_w, Va_b, scores);

    softmax_ctx_kernel<<<dim3(B_), dim3(256), 0, stream>>>(keys, scores, outC, outW);
}
